// FPAC_36670430773909
// MI455X (gfx1250) — matrix operand order repaired and checked by reading, not compiled, not run on hardware
//
#include <hip/hip_runtime.h>
#include <hip/hip_bf16.h>

#define B_    4
#define N_    1024
#define C61   61
#define S_    256
#define NS    32
#define V_    27
#define CIN   64
#define COUT  128
#define MID_  16
#define BSN   (B_*S_*NS)        /* 32768 */
#define R_TOT (BSN + V_)        /* 32795 */
#define RV    (R_TOT * V_)      /* 885465 */
#define RVPAD 885504            /* ceil(RV/128)*128 */
#define EPS_  1e-5f

typedef _Float16 v16h __attribute__((ext_vector_type(16)));
typedef _Float16 v8h  __attribute__((ext_vector_type(8)));
typedef float    v8f  __attribute__((ext_vector_type(8)));

// ---------------------------------------------------------------------------
// Weight transpose+convert: W[co][ci] f32 -> Wt[ci][co] f16
// ---------------------------------------------------------------------------
__global__ void k_wconv(const float* __restrict__ W, _Float16* __restrict__ Wt,
                        int co, int ci) {
  int t = blockIdx.x * 256 + threadIdx.x;
  if (t < co * ci) {
    int o = t / ci, k = t - o * ci;
    Wt[k * co + o] = (_Float16)W[t];
  }
}

// ---------------------------------------------------------------------------
// Farthest point sampling: one block (1024 thr) per batch, serial over S.
// Records incoming `far` each step (matches the jax scan semantics).
// ---------------------------------------------------------------------------
__global__ __launch_bounds__(1024)
void k_fps(const float* __restrict__ xyz, int* __restrict__ fpsidx,
           float* __restrict__ nxz, float* __restrict__ out_tail) {
  int b = blockIdx.x, tid = threadIdx.x;
  __shared__ float sd[1024];
  __shared__ int   si[1024];
  const float* X = xyz + (long)b * N_ * 3;
  float px = X[tid*3], py = X[tid*3+1], pz = X[tid*3+2];
  float mind = 1e10f;
  int far = 0;
  for (int s = 0; s < S_; s++) {
    if (tid == 0) {
      fpsidx[b*S_ + s] = far;
      float x0 = X[far*3], x1 = X[far*3+1], x2 = X[far*3+2];
      float* nw = nxz + (b*S_ + s)*3;
      nw[0]=x0; nw[1]=x1; nw[2]=x2;
      float* ot = out_tail + (b*S_ + s)*3;
      ot[0]=x0; ot[1]=x1; ot[2]=x2;
    }
    float cx = X[far*3], cy = X[far*3+1], cz = X[far*3+2];
    float dx = px-cx, dy = py-cy, dz = pz-cz;
    mind = fminf(mind, dx*dx + dy*dy + dz*dz);
    sd[tid] = mind; si[tid] = tid;
    __syncthreads();
    for (int off = 512; off > 0; off >>= 1) {
      if (tid < off) {
        float o2 = sd[tid+off];
        if (o2 > sd[tid] || (o2 == sd[tid] && si[tid+off] < si[tid])) {
          sd[tid] = o2; si[tid] = si[tid+off];
        }
      }
      __syncthreads();
    }
    far = si[0];
    __syncthreads();
  }
}

// ---------------------------------------------------------------------------
// Ball query + gather. One wave per (b,s). Ascending-index first-32 selection
// via wave32 ballot + prefix popcount. Writes raw gx (for m1 input) and
// masked gf [slot][64] (3 rel-xyz + 61 feats).
// ---------------------------------------------------------------------------
__global__ void k_ballgather(const float* __restrict__ xyz, const float* __restrict__ f,
                             const float* __restrict__ nxz,
                             float* __restrict__ gx, float* __restrict__ gf,
                             int* __restrict__ maskb) {
  int bs = blockIdx.x, lane = threadIdx.x;
  int b = bs >> 8;
  const float* X = xyz + (long)b * N_ * 3;
  const float* F = f   + (long)b * N_ * C61;
  float cx = nxz[bs*3], cy = nxz[bs*3+1], cz = nxz[bs*3+2];
  __shared__ int sidx[NS];
  const float rr = 0.2f * 0.2f;
  int cnt = 0;
  for (int base = 0; base < N_; base += 32) {
    int j = base + lane;
    float dx = X[j*3]-cx, dy = X[j*3+1]-cy, dz = X[j*3+2]-cz;
    float sq = dx*dx + dy*dy + dz*dz;
    bool pred = !(sq > rr);
    unsigned mm = (unsigned)__ballot(pred);
    if (pred) {
      int pos = cnt + __popc(mm & ((1u << lane) - 1u));
      if (pos < NS) sidx[pos] = j;
    }
    cnt += __popc(mm);
    if (cnt >= NS) break;
  }
  __syncthreads();
  int valid = cnt < NS ? cnt : NS;
  int i = lane;
  bool msk = (i >= valid);
  int idx = msk ? sidx[0] : sidx[i];
  long row = (long)bs * NS + i;
  float g0 = X[idx*3]-cx, g1 = X[idx*3+1]-cy, g2 = X[idx*3+2]-cz;
  gx[row*3] = g0; gx[row*3+1] = g1; gx[row*3+2] = g2;
  float* grow = gf + row * CIN;
  grow[0] = msk ? 0.f : g0;
  grow[1] = msk ? 0.f : g1;
  grow[2] = msk ? 0.f : g2;
  for (int c = 0; c < C61; c++) grow[3+c] = msk ? 0.f : F[(long)idx*C61 + c];
  maskb[row] = msk ? 1 : 0;
}

// ---------------------------------------------------------------------------
// m1 layer1 (6->32) on the fly. mode 0: accumulate BN stats; mode 1: write
// normalized+ReLU f16 activations x1[p][32].
// ---------------------------------------------------------------------------
__global__ void k_m1pass(const float* __restrict__ gx, const float* __restrict__ fp,
                         const float* __restrict__ W1, const float* __restrict__ b1,
                         const float* __restrict__ scale, const float* __restrict__ shift,
                         float* sums, float* sumsq, _Float16* __restrict__ x1, int mode) {
  __shared__ float ls[32], lq[32];
  int tid = threadIdx.x;
  if (mode == 0 && tid < 32) { ls[tid] = 0.f; lq[tid] = 0.f; }
  __syncthreads();
  long p = (long)blockIdx.x * blockDim.x + tid;
  if (p < (long)RV) {
    int r = (int)(p / V_);
    int v = (int)(p - (long)r * V_);
    float in6[6];
    if (r < BSN) { in6[0]=gx[r*3]; in6[1]=gx[r*3+1]; in6[2]=gx[r*3+2]; }
    else { int q = r - BSN; in6[0]=fp[q*3]; in6[1]=fp[q*3+1]; in6[2]=fp[q*3+2]; }
    in6[3]=fp[v*3]; in6[4]=fp[v*3+1]; in6[5]=fp[v*3+2];
    for (int o = 0; o < 32; o++) {
      float acc = b1[o];
      #pragma unroll
      for (int k = 0; k < 6; k++) acc += W1[o*6+k] * in6[k];
      if (mode == 0) { atomicAdd(&ls[o], acc); atomicAdd(&lq[o], acc*acc); }
      else {
        float y = acc * scale[o] + shift[o];
        x1[p*32 + o] = (_Float16)(y > 0.f ? y : 0.f);
      }
    }
  }
  if (mode == 0) {
    __syncthreads();
    if (tid < 32) { atomicAdd(&sums[tid], ls[tid]); atomicAdd(&sumsq[tid], lq[tid]); }
  }
}

// ---------------------------------------------------------------------------
// WMMA GEMM: Y[M,N] = A[M,KT] * Bw[KT,N] + bias, f16 in, f32 acc, f16 out.
// Also accumulates per-channel sum/sumsq for batch-norm.
// Block = 256 thr (8 waves). Each wave owns a 16-row M tile; block owns one
// 16-col N tile. The B column block is staged in LDS *pre-swizzled into the
// per-lane WMMA fragment order*, so each lane loads its whole B fragment as
// one contiguous 32B read (2x ds_load_b128) per K-step.
// ---------------------------------------------------------------------------
template <int KT>
__global__ __launch_bounds__(256)
void wmma_gemm(const _Float16* __restrict__ A, const _Float16* __restrict__ Bw,
               const float* __restrict__ bias, _Float16* __restrict__ Yout,
               float* sums, float* sumsq, int Mvalid, int Ncols) {
  // Fragment order: for absolute k, kin=k&31, block=k>>5,
  //   lane = ((kin>>3)&1)*16 + nn,  half = (kin&7) | ((kin>>4)<<3)
  // Bs index = block*512 + lane*16 + half
  __shared__ __align__(32) _Float16 Bs[KT * 16];
  __shared__ float ls[16], lq[16];
  int tid = threadIdx.x;
  int n0 = blockIdx.y * 16;
  for (int i = tid; i < KT * 16; i += 256) {
    int k = i >> 4, nn = i & 15;
    int kin = k & 31;
    int dst = (k >> 5) * 512 + ((((kin >> 3) & 1) << 4) + nn) * 16 + ((kin & 7) | ((kin >> 4) << 3));
    Bs[dst] = Bw[k * Ncols + n0 + nn];
  }
  if (tid < 16) { ls[tid] = 0.f; lq[tid] = 0.f; }
  __syncthreads();

  int wave = tid >> 5, lane = tid & 31;
  int hi = lane >> 4, low = lane & 15;
  int mBase = blockIdx.x * 128 + wave * 16;
  int rowA = mBase + low;
  bool rowok = rowA < Mvalid;
  const _Float16* Arow = A + (long)rowA * KT;
  const v16h* Bfrag = (const v16h*)(Bs + lane * 16);   // stride 32 v16h per k-block

  v8f acc = {0.f,0.f,0.f,0.f,0.f,0.f,0.f,0.f};
  for (int k0 = 0; k0 < KT; k0 += 32) {
    v16h a = {};
    if (rowok) {
      v8h a0 = *(const v8h*)(Arow + k0 + hi * 8);        // K = k0+hi*8 .. +7
      v8h a1 = *(const v8h*)(Arow + k0 + 16 + hi * 8);   // K = k0+16+hi*8 .. +7
      #pragma unroll
      for (int t = 0; t < 8; t++) { a[t] = a0[t]; a[t+8] = a1[t]; }
    }
    v16h bv = Bfrag[(k0 >> 5) * 32];
    acc = __builtin_amdgcn_wmma_f32_16x16x32_f16(false, a, false, bv,
                                                 (short)0, acc, false, false);
  }

  int ch = n0 + low;
  float bsv = bias ? bias[ch] : 0.f;
  float psum = 0.f, psq = 0.f;
  if (mBase + 16 <= Mvalid) {
    // fast path: whole tile valid, unguarded strided stores
    _Float16* Ycol = Yout + (long)(mBase + hi * 8) * Ncols + ch;
    #pragma unroll
    for (int rI = 0; rI < 8; rI++) {
      float val = acc[rI] + bsv;
      Ycol[(long)rI * Ncols] = (_Float16)val;
      psum += val; psq += val * val;
    }
  } else {
    #pragma unroll
    for (int rI = 0; rI < 8; rI++) {
      int rowO = mBase + hi * 8 + rI;        // D layout: m = (lane>=16)*8 + vgpr
      float val = acc[rI] + bsv;
      if (rowO < Mvalid) {
        Yout[(long)rowO * Ncols + ch] = (_Float16)val;
        psum += val; psq += val * val;
      }
    }
  }
  if (sums) {
    atomicAdd(&ls[low], psum);
    atomicAdd(&lq[low], psq);
    __syncthreads();
    if (tid < 16) { atomicAdd(&sums[n0 + tid], ls[tid]); atomicAdd(&sumsq[n0 + tid], lq[tid]); }
  }
}

// ---------------------------------------------------------------------------
// BN finalize: scale = g*rsqrt(var+eps); shift = be - mean*scale
// ---------------------------------------------------------------------------
__global__ void k_finalize(const float* __restrict__ sums, const float* __restrict__ sumsq,
                           const float* __restrict__ g, const float* __restrict__ be,
                           float* __restrict__ scale, float* __restrict__ shift,
                           int Cc, float invN) {
  int c = threadIdx.x;
  if (c < Cc) {
    float m = sums[c] * invN;
    float var = sumsq[c] * invN - m * m;
    float sc = g[c] * rsqrtf(var + EPS_);
    scale[c] = sc;
    shift[c] = be[c] - m * sc;
  }
}

// In-place f16 BN + ReLU (Cc is a power of two)
__global__ void k_bnapply(_Float16* __restrict__ x, const float* __restrict__ scale,
                          const float* __restrict__ shift, long total, int Cc) {
  long t = (long)blockIdx.x * 256 + threadIdx.x;
  if (t < total) {
    int c = (int)(t & (Cc - 1));
    float v = (float)x[t] * scale[c] + shift[c];
    x[t] = (_Float16)(v > 0.f ? v : 0.f);
  }
}

// ---------------------------------------------------------------------------
// m2 branch (tiny): layer A 8192->256 over length 27, BN over length, ReLU
// ---------------------------------------------------------------------------
__global__ void k_m2a(const float* __restrict__ w, const float* __restrict__ WA,
                      const float* __restrict__ bA, const float* __restrict__ gA,
                      const float* __restrict__ beA, float* __restrict__ hm2) {
  int o = blockIdx.x, lane = threadIdx.x;
  __shared__ float sy[V_];
  float y = 0.f;
  if (lane < V_) {
    const float* wr = w + (long)lane * (CIN * COUT);
    const float* Wr = WA + (long)o * (CIN * COUT);
    for (int c = 0; c < CIN * COUT; c++) y += Wr[c] * wr[c];
    y += bA[o];
    sy[lane] = y;
  }
  __syncthreads();
  float m = 0.f, q = 0.f;
  for (int i = 0; i < V_; i++) m += sy[i];
  m /= (float)V_;
  for (int i = 0; i < V_; i++) { float d = sy[i] - m; q += d * d; }
  q /= (float)V_;
  if (lane < V_) {
    float val = (y - m) * rsqrtf(q + EPS_) * gA[o] + beA[o];
    hm2[o * V_ + lane] = val > 0.f ? val : 0.f;
  }
}

// m2 layer B 256->16, BN over length, ReLU; writes w2arr[v][mid]
__global__ void k_m2b(const float* __restrict__ hm2, const float* __restrict__ WB,
                      const float* __restrict__ bB, const float* __restrict__ gB,
                      const float* __restrict__ beB, float* __restrict__ w2arr) {
  __shared__ float sy[MID_ * V_];
  __shared__ float ssc[MID_], ssh[MID_];
  int t = threadIdx.x;
  if (t < MID_ * V_) {
    int mid = t / V_, v = t - mid * V_;
    float y = bB[mid];
    for (int k = 0; k < 256; k++) y += WB[mid * 256 + k] * hm2[k * V_ + v];
    sy[mid * V_ + v] = y;
  }
  __syncthreads();
  if (t < MID_) {
    float m = 0.f, q = 0.f;
    for (int v = 0; v < V_; v++) m += sy[t * V_ + v];
    m /= (float)V_;
    for (int v = 0; v < V_; v++) { float d = sy[t * V_ + v] - m; q += d * d; }
    q /= (float)V_;
    ssc[t] = rsqrtf(q + EPS_) * gB[t];
    ssh[t] = beB[t] - m * ssc[t];
  }
  __syncthreads();
  if (t < MID_ * V_) {
    int mid = t / V_, v = t - mid * V_;
    float val = sy[mid * V_ + v] * ssc[mid] + ssh[mid];
    w2arr[v * MID_ + mid] = val > 0.f ? val : 0.f;
  }
}

// ws[r][mid] = sum_v relu(bn(y3[(r*V+v)][mid])) * w2[v][mid]
__global__ void k_ws(const _Float16* __restrict__ y3, const float* __restrict__ sc3,
                     const float* __restrict__ sh3, const float* __restrict__ w2arr,
                     float* __restrict__ wsArr) {
  int t = blockIdx.x * 256 + threadIdx.x;
  if (t >= R_TOT * MID_) return;
  int r = t >> 4, mid = t & 15;
  float sc = sc3[mid], sh = sh3[mid];
  float s = 0.f;
  for (int v = 0; v < V_; v++) {
    float val = (float)y3[((long)(r * V_ + v)) * MID_ + mid] * sc + sh;
    val = val > 0.f ? val : 0.f;
    s += val * w2arr[v * MID_ + mid];
  }
  wsArr[t] = s;
}

// out_pre[bs][c*16+m] = sum_i gf[bs,i,c] * wp[bs,i,m]   (wp = mask? 0 : ws)
__global__ void k_einsum(const float* __restrict__ gf, const float* __restrict__ wsArr,
                         const int* __restrict__ maskb, _Float16* __restrict__ outPre) {
  __shared__ float sgf[NS * CIN];
  __shared__ float swp[NS * MID_];
  int bs = blockIdx.x, tid = threadIdx.x;
  long gbase = (long)bs * NS * CIN;
  for (int i = tid; i < NS * CIN; i += 256) sgf[i] = gf[gbase + i];
  for (int i = tid; i < NS * MID_; i += 256) {
    int slot = i >> 4, mid = i & 15;
    int row = bs * NS + slot;
    swp[i] = maskb[row] ? 0.f : wsArr[row * MID_ + mid];
  }
  __syncthreads();
  for (int oc = tid; oc < CIN * MID_; oc += 256) {
    int c = oc >> 4, m = oc & 15;
    float s = 0.f;
    #pragma unroll 8
    for (int i = 0; i < NS; i++) s += sgf[i * CIN + c] * swp[i * MID_ + m];
    outPre[(long)bs * (CIN * MID_) + oc] = (_Float16)s;
  }
}

// final: out[b][co][s] (flat, reinterpreted [B,S,cout]) = relu(bn(h2[(b*S+s)][co]))
__global__ void k_final(const _Float16* __restrict__ h2, const float* __restrict__ sc,
                        const float* __restrict__ sh, float* __restrict__ out) {
  int t = blockIdx.x * 256 + threadIdx.x;
  if (t >= B_ * COUT * S_) return;
  int b = t >> 15;
  int rem = t & 32767;
  int co = rem >> 8;
  int s = rem & 255;
  float v = (float)h2[((long)(b * S_ + s)) * COUT + co] * sc[co] + sh[co];
  out[t] = v > 0.f ? v : 0.f;
}

// ---------------------------------------------------------------------------
extern "C" void kernel_launch(void* const* d_in, const int* in_sizes, int n_in,
                              void* d_out, int out_size, void* d_ws, size_t ws_size,
                              hipStream_t stream) {
  (void)in_sizes; (void)n_in; (void)out_size;
  const float* xyz = (const float*)d_in[0];
  const float* f   = (const float*)d_in[1];
  const float* w   = (const float*)d_in[2];
  const float* fp  = (const float*)d_in[3];
  const float *m1W1=(const float*)d_in[4],  *m1b1=(const float*)d_in[5],
              *m1g1=(const float*)d_in[6],  *m1e1=(const float*)d_in[7];
  const float *m1W2=(const float*)d_in[8],  *m1b2=(const float*)d_in[9],
              *m1g2=(const float*)d_in[10], *m1e2=(const float*)d_in[11];
  const float *m1W3=(const float*)d_in[12], *m1b3=(const float*)d_in[13],
              *m1g3=(const float*)d_in[14], *m1e3=(const float*)d_in[15];
  const float *m2WA=(const float*)d_in[16], *m2bA=(const float*)d_in[17],
              *m2gA=(const float*)d_in[18], *m2eA=(const float*)d_in[19];
  const float *m2WB=(const float*)d_in[20], *m2bB=(const float*)d_in[21],
              *m2gB=(const float*)d_in[22], *m2eB=(const float*)d_in[23];
  const float *mrW1=(const float*)d_in[24], *mrb1=(const float*)d_in[25],
              *mrg1=(const float*)d_in[26], *mre1=(const float*)d_in[27];
  const float *mrW2=(const float*)d_in[28], *mrb2=(const float*)d_in[29],
              *mrg2=(const float*)d_in[30], *mre2=(const float*)d_in[31];
  float* outF = (float*)d_out;

  size_t off = 0;
  auto alloc = [&](size_t bytes) -> char* {
    char* p = (char*)d_ws + off;
    off += (bytes + 255) & ~(size_t)255;
    return p;
  };
  _Float16* x1     = (_Float16*)alloc((size_t)RVPAD * 32 * 2);
  _Float16* y2     = (_Float16*)alloc((size_t)RVPAD * 64 * 2);
  _Float16* y3     = (_Float16*)alloc((size_t)RVPAD * 16 * 2);
  float*    wsArr  = (float*)   alloc((size_t)R_TOT * MID_ * 4);
  float*    gx     = (float*)   alloc((size_t)BSN * 3 * 4);
  float*    gf     = (float*)   alloc((size_t)BSN * CIN * 4);
  int*      maskb  = (int*)     alloc((size_t)BSN * 4);
  int*      fpsidx = (int*)     alloc((size_t)B_ * S_ * 4);
  float*    nxz    = (float*)   alloc((size_t)B_ * S_ * 3 * 4);
  _Float16* outPre = (_Float16*)alloc((size_t)1024 * 1024 * 2);
  _Float16* h1     = (_Float16*)alloc((size_t)1024 * 256 * 2);
  _Float16* h2     = (_Float16*)alloc((size_t)1024 * 128 * 2);
  _Float16* W2t    = (_Float16*)alloc((size_t)32 * 64 * 2);
  _Float16* W3t    = (_Float16*)alloc((size_t)64 * 16 * 2);
  _Float16* Wr1t   = (_Float16*)alloc((size_t)1024 * 256 * 2);
  _Float16* Wr2t   = (_Float16*)alloc((size_t)256 * 128 * 2);
  float*    hm2    = (float*)   alloc((size_t)256 * V_ * 4);
  float*    w2arr  = (float*)   alloc((size_t)V_ * MID_ * 4);
  float*    stats  = (float*)   alloc((size_t)992 * 4);
  float*    scsh   = (float*)   alloc((size_t)992 * 4);
  if (off > ws_size) return;

  float *sum1=stats,      *sq1=stats+32,
        *sum2=stats+64,   *sq2=stats+128,
        *sum3=stats+192,  *sq3=stats+208,
        *sum4=stats+224,  *sq4=stats+480,
        *sum5=stats+736,  *sq5=stats+864;
  float *sc1=scsh,        *sh1=scsh+32,
        *sc2=scsh+64,     *sh2=scsh+128,
        *sc3=scsh+192,    *sh3=scsh+208,
        *sc4=scsh+224,    *sh4=scsh+480,
        *sc5=scsh+736,    *sh5=scsh+864;

  // weight transpose+f16
  k_wconv<<<(64*32+255)/256,   256, 0, stream>>>(m1W2, W2t, 64, 32);
  k_wconv<<<(16*64+255)/256,   256, 0, stream>>>(m1W3, W3t, 16, 64);
  k_wconv<<<(256*1024+255)/256,256, 0, stream>>>(mrW1, Wr1t, 256, 1024);
  k_wconv<<<(128*256+255)/256, 256, 0, stream>>>(mrW2, Wr2t, 128, 256);

  // sampling + grouping
  k_fps<<<B_, 1024, 0, stream>>>(xyz, fpsidx, nxz, outF + B_*S_*COUT);
  k_ballgather<<<B_*S_, 32, 0, stream>>>(xyz, f, nxz, gx, gf, maskb);

  hipMemsetAsync(stats, 0, 992 * 4, stream);

  const int rvBlocks = (RV + 255) / 256;
  const float invRV = 1.0f / (float)RV;
  const float invBS = 1.0f / 1024.0f;

  // m1 layer 1 (scalar, K=6): stats then normalized f16 write
  k_m1pass<<<rvBlocks, 256, 0, stream>>>(gx, fp, m1W1, m1b1, sc1, sh1, sum1, sq1, x1, 0);
  k_finalize<<<1, 32, 0, stream>>>(sum1, sq1, m1g1, m1e1, sc1, sh1, 32, invRV);
  k_m1pass<<<rvBlocks, 256, 0, stream>>>(gx, fp, m1W1, m1b1, sc1, sh1, sum1, sq1, x1, 1);

  // m1 layer 2 (WMMA, K=32, N=64)
  wmma_gemm<32><<<dim3(RVPAD/128, 4), 256, 0, stream>>>(x1, W2t, m1b2, y2, sum2, sq2, RV, 64);
  k_finalize<<<1, 64, 0, stream>>>(sum2, sq2, m1g2, m1e2, sc2, sh2, 64, invRV);
  k_bnapply<<<(int)(((long)RV*64 + 255)/256), 256, 0, stream>>>(y2, sc2, sh2, (long)RV*64, 64);

  // m1 layer 3 (WMMA, K=64, N=16); BN applied on the fly in k_ws
  wmma_gemm<64><<<dim3(RVPAD/128, 1), 256, 0, stream>>>(y2, W3t, m1b3, y3, sum3, sq3, RV, 16);
  k_finalize<<<1, 16, 0, stream>>>(sum3, sq3, m1g3, m1e3, sc3, sh3, 16, invRV);

  // m2 branch -> w2arr[v][mid]
  k_m2a<<<256, 32, 0, stream>>>(w, m2WA, m2bA, m2gA, m2eA, hm2);
  k_m2b<<<1, 512, 0, stream>>>(hm2, m2WB, m2bB, m2gB, m2eB, w2arr);

  // ws reduction over V, then masked per-center einsum -> out_pre f16
  k_ws<<<(R_TOT*MID_ + 255)/256, 256, 0, stream>>>(y3, sc3, sh3, w2arr, wsArr);
  k_einsum<<<B_*S_, 256, 0, stream>>>(gf, wsArr, maskb, outPre);

  // mr layer 1 (WMMA, K=1024, N=256)
  wmma_gemm<1024><<<dim3(8, 16), 256, 0, stream>>>(outPre, Wr1t, mrb1, h1, sum4, sq4, 1024, 256);
  k_finalize<<<1, 256, 0, stream>>>(sum4, sq4, mrg1, mre1, sc4, sh4, 256, invBS);
  k_bnapply<<<(1024*256 + 255)/256, 256, 0, stream>>>(h1, sc4, sh4, (long)1024*256, 256);

  // mr layer 2 (WMMA, K=256, N=128)
  wmma_gemm<256><<<dim3(8, 8), 256, 0, stream>>>(h1, Wr2t, mrb2, h2, sum5, sq5, 1024, 128);
  k_finalize<<<1, 128, 0, stream>>>(sum5, sq5, mrg2, mre2, sc5, sh5, 128, invBS);

  // final BN+ReLU with the [B,cout,S] -> [B,S,cout] reinterpret mapping
  k_final<<<(B_*COUT*S_ + 255)/256, 256, 0, stream>>>(h2, sc5, sh5, outF);
}
